// RN2_54820962566527
// MI455X (gfx1250) — matrix operand order repaired and checked by reading, not compiled, not run on hardware
//
#include <hip/hip_runtime.h>

typedef __attribute__((ext_vector_type(16))) _Float16 v16h;
typedef __attribute__((ext_vector_type(8)))  float    v8f;
typedef __attribute__((ext_vector_type(4)))  float    f4;   // native vector for nontemporal builtins

#define BB    32
#define MM    1024
#define VDIM  1024
#define QDIM  1024
#define HID   310
#define HIDP  320   // HID padded to multiple of 32 (WMMA K-step)

// ---------------------------------------------------------------------------
// Kernel 1: gating MLPs with v_wmma_f32_16x16x32_f16.
//   grid.x = 2 (gate 1 / gate 2), block = 256 threads = 8 waves (wave32).
//   GEMM1: H[32x310] = relu(q[32x1024] @ Wc^T + bc)   -> LDS (f16, zero-padded)
//   GEMM2: G[32x1024] = sigmoid(H @ We^T + be)        -> global ws
// ---------------------------------------------------------------------------
__global__ __launch_bounds__(256) void rn2_gate(
    const float* __restrict__ q,
    const float* __restrict__ Wc1, const float* __restrict__ bc1,
    const float* __restrict__ We1, const float* __restrict__ be1,
    const float* __restrict__ Wc2, const float* __restrict__ bc2,
    const float* __restrict__ We2, const float* __restrict__ be2,
    float* __restrict__ q1o, float* __restrict__ q2o)
{
    const int tid  = threadIdx.x;
    const int wave = tid >> 5;
    const int lane = tid & 31;
    const int hi   = lane >> 4;   // lane half: 0 or 1
    const int lm   = lane & 15;   // row (A) / col (B,D) within 16x16 tile

    const float* Wc  = blockIdx.x ? Wc2 : Wc1;
    const float* bc  = blockIdx.x ? bc2 : bc1;
    const float* We  = blockIdx.x ? We2 : We1;
    const float* be  = blockIdx.x ? be2 : be1;
    float*       gout = blockIdx.x ? q2o : q1o;

    __shared__ _Float16 Hs[BB][HIDP];
    for (int i = tid; i < BB * HIDP; i += 256) ((_Float16*)Hs)[i] = (_Float16)0.f;
    __syncthreads();

    // -------- GEMM1: 2 M-tiles x 20 N-tiles, K = 1024 --------------------
    for (int tile = wave; tile < 2 * 20; tile += 8) {      // 40/8 = 5 per wave (uniform)
        const int m0 = (tile / 20) * 16;
        const int n0 = (tile % 20) * 16;
        const int am = m0 + lm;          // A-matrix row this lane supplies
        const int bn = n0 + lm;          // B-matrix column this lane supplies
        v8f c = {0.f,0.f,0.f,0.f,0.f,0.f,0.f,0.f};
        for (int k0 = 0; k0 < QDIM; k0 += 32) {
            v16h a, b;
            // A operand, contiguous k order chosen here (same as B): lane g*16+m, elem i -> k=g*16+i
            const float* qa = q + am * QDIM + k0 + hi * 16;
            #pragma unroll
            for (int i = 0; i < 8; ++i) { a[i] = (_Float16)qa[i]; a[i + 8] = (_Float16)qa[8 + i]; }
            // B operand, same contiguous k order as A: lane g*16+n, elem i -> k=g*16+i ; B[k][n] = Wc[n][k]
            if (bn < HID) {
                const float* wp = Wc + bn * QDIM + k0 + hi * 16;
                #pragma unroll
                for (int i = 0; i < 16; ++i) b[i] = (_Float16)wp[i];
            } else {
                #pragma unroll
                for (int i = 0; i < 16; ++i) b[i] = (_Float16)0.f;
            }
            c = __builtin_amdgcn_wmma_f32_16x16x32_f16(false, a, false, b, (short)0, c, false, false);
        }
        // D layout: VGPR r, lane g*16+n -> D[m0 + 8*g + r][n0 + n]
        if (bn < HID) {
            const float bias = bc[bn];
            #pragma unroll
            for (int r = 0; r < 8; ++r) {
                float h = c[r] + bias;
                Hs[m0 + 8 * hi + r][bn] = (_Float16)(h > 0.f ? h : 0.f);
            }
        }
    }
    __syncthreads();

    // -------- GEMM2: 2 M-tiles x 64 N-tiles, K = 320 (padded) ------------
    for (int tile = wave; tile < 2 * 64; tile += 8) {      // 128/8 = 16 per wave (uniform)
        const int m0 = (tile >> 6) * 16;
        const int n0 = (tile & 63) * 16;
        const int am = m0 + lm;
        const int bn = n0 + lm;
        v8f c = {0.f,0.f,0.f,0.f,0.f,0.f,0.f,0.f};
        for (int k0 = 0; k0 < HIDP; k0 += 32) {
            v16h a, b;
            const _Float16* ha = &Hs[am][k0 + hi * 16];
            #pragma unroll
            for (int i = 0; i < 8; ++i) { a[i] = ha[i]; a[i + 8] = ha[8 + i]; }
            const int kb = k0 + hi * 16;
            const float* wp = We + bn * HID + kb;          // We is (1024, 310) row-major
            #pragma unroll
            for (int i = 0; i < 16; ++i) b[i] = (kb + i < HID) ? (_Float16)wp[i] : (_Float16)0.f;
            c = __builtin_amdgcn_wmma_f32_16x16x32_f16(false, a, false, b, (short)0, c, false, false);
        }
        const float bias = be[bn];
        #pragma unroll
        for (int r = 0; r < 8; ++r) {
            float x = c[r] + bias;
            gout[(m0 + 8 * hi + r) * VDIM + bn] = 1.f / (1.f + __expf(-x));
        }
    }
}

// ---------------------------------------------------------------------------
// Kernel 2: partial column sums of v. grid (32 b, 32 splits), 256 thr x float4
// covers d = 0..1023 exactly. Reads v once (warms 192MB L2 with the 128MB v).
// ---------------------------------------------------------------------------
__global__ __launch_bounds__(256) void rn2_colsum(const float* __restrict__ v,
                                                  float* __restrict__ partial)
{
    const int b  = blockIdx.x;
    const int sp = blockIdx.y;
    const int d  = threadIdx.x * 4;
    const f4* vp = (const f4*)(v + ((size_t)b * MM + (size_t)sp * 32) * VDIM + d);
    f4 acc = {0.f, 0.f, 0.f, 0.f};
    #pragma unroll 4
    for (int r = 0; r < 32; ++r) {
        acc += vp[(size_t)r * (VDIM / 4)];
    }
    *(f4*)(partial + ((size_t)(b * 32 + sp)) * VDIM + d) = acc;
}

// ---------------------------------------------------------------------------
// Kernel 3: reduce partials, fold gates:  a = q1*vsum,  s = m*q2.
// ---------------------------------------------------------------------------
__global__ __launch_bounds__(256) void rn2_coeff(const float* __restrict__ partial,
                                                 const float* __restrict__ q1,
                                                 const float* __restrict__ q2,
                                                 float* __restrict__ a,
                                                 float* __restrict__ s)
{
    const int b = blockIdx.x;
    const int d = threadIdx.x * 4;
    const f4* pp = (const f4*)(partial + (size_t)b * 32 * VDIM + d);
    f4 acc = {0.f, 0.f, 0.f, 0.f};
    #pragma unroll 4
    for (int r = 0; r < 32; ++r) {
        acc += pp[(size_t)r * (VDIM / 4)];
    }
    f4 g1 = *(const f4*)(q1 + b * VDIM + d);
    f4 g2 = *(const f4*)(q2 + b * VDIM + d);
    *(f4*)(a + b * VDIM + d) = g1 * acc;
    *(f4*)(s + b * VDIM + d) = g2 * (float)MM;
}

// ---------------------------------------------------------------------------
// Kernel 4: out[b,j,d] = a[b,d] + s[b,d]*v[b,j,d]. grid (32 b, 64 j-chunks of
// 16). v loads should hit L2 (v fits in 192MB); out uses non-temporal stores
// so the 128MB output stream does not evict v from L2.
// ---------------------------------------------------------------------------
__global__ __launch_bounds__(256) void rn2_out(const float* __restrict__ v,
                                               const float* __restrict__ a,
                                               const float* __restrict__ s,
                                               float* __restrict__ out)
{
    const int b  = blockIdx.x;
    const int jc = blockIdx.y;        // 64 chunks of 16 rows
    const int d  = threadIdx.x * 4;
    f4 a4 = *(const f4*)(a + b * VDIM + d);
    f4 s4 = *(const f4*)(s + b * VDIM + d);
    const size_t base = ((size_t)b * MM + (size_t)jc * 16) * VDIM + d;
    const f4* vp = (const f4*)(v + base);
    f4*       op = (f4*)(out + base);
    #pragma unroll 4
    for (int r = 0; r < 16; ++r) {
        f4 x = vp[(size_t)r * (VDIM / 4)];
        f4 o = a4 + s4 * x;
        __builtin_nontemporal_store(o, op + (size_t)r * (VDIM / 4));
    }
}

// ---------------------------------------------------------------------------
extern "C" void kernel_launch(void* const* d_in, const int* in_sizes, int n_in,
                              void* d_out, int out_size, void* d_ws, size_t ws_size,
                              hipStream_t stream)
{
    (void)in_sizes; (void)n_in; (void)out_size; (void)ws_size;
    const float* v   = (const float*)d_in[0];
    const float* q   = (const float*)d_in[1];
    const float* Wc1 = (const float*)d_in[2];
    const float* bc1 = (const float*)d_in[3];
    const float* We1 = (const float*)d_in[4];
    const float* be1 = (const float*)d_in[5];
    const float* Wc2 = (const float*)d_in[6];
    const float* bc2 = (const float*)d_in[7];
    const float* We2 = (const float*)d_in[8];
    const float* be2 = (const float*)d_in[9];
    float* out = (float*)d_out;

    float* ws      = (float*)d_ws;
    float* q1o     = ws;                       // 32*1024
    float* q2o     = ws + 1 * BB * VDIM;       // 32*1024
    float* ac      = ws + 2 * BB * VDIM;       // 32*1024
    float* sc      = ws + 3 * BB * VDIM;       // 32*1024
    float* partial = ws + 4 * BB * VDIM;       // 32*32*1024 (4MB)

    rn2_gate<<<dim3(2), 256, 0, stream>>>(q, Wc1, bc1, We1, be1,
                                          Wc2, bc2, We2, be2, q1o, q2o);
    rn2_colsum<<<dim3(BB, 32), 256, 0, stream>>>(v, partial);
    rn2_coeff<<<dim3(BB), 256, 0, stream>>>(partial, q1o, q2o, ac, sc);
    rn2_out<<<dim3(BB, 64), 256, 0, stream>>>(v, ac, sc, out);
}
